// SAI2E_77996606095744
// MI455X (gfx1250) — matrix operand order repaired and checked by reading, not compiled, not run on hardware
//
#include <hip/hip_runtime.h>
#include <math.h>

typedef __attribute__((ext_vector_type(8)))  __bf16 v8bf;
typedef __attribute__((ext_vector_type(16))) __bf16 v16bf;
typedef __attribute__((ext_vector_type(8)))  float  v8f;
typedef unsigned int u32x4 __attribute__((ext_vector_type(4)));
typedef int          i32x8 __attribute__((ext_vector_type(8)));
typedef int          i32x4 __attribute__((ext_vector_type(4)));

union V16 { v16bf v; v8bf h[2]; };

#ifndef __has_builtin
#define __has_builtin(x) 0
#endif
#if __has_builtin(__builtin_amdgcn_tensor_load_to_lds) && \
    __has_builtin(__builtin_amdgcn_s_wait_tensorcnt)
#define USE_TDM 1
#else
#define USE_TDM 0
#endif

__device__ __forceinline__ float gelu_exact(float v) {
    return 0.5f * v * (1.0f + erff(v * 0.70710678118654752f));
}
__device__ __forceinline__ int iclamp(int v, int lo, int hi) {
    return v < lo ? lo : (v > hi ? hi : v);
}

// ---------------------------------------------------------------------------
// Kernel 1: convert fp32 weights -> bf16 in WMMA B layouts.
//   bg1a/bg1b : [48][448], K = (dy*3+dx)*48 + ci, K in [432,448) zero pad
//   bg2a      : [16][64],  rows >=4 zero, K >= 48 zero
//   bg2b      : [48][64],  K >= 48 zero
// ---------------------------------------------------------------------------
__global__ __launch_bounds__(256) void wconv_kernel(
    const float* __restrict__ w1_off, const float* __restrict__ w1_mod,
    const float* __restrict__ w2_off, const float* __restrict__ w2_mod,
    __bf16* __restrict__ bg1a, __bf16* __restrict__ bg1b,
    __bf16* __restrict__ bg2a, __bf16* __restrict__ bg2b) {
    const int stride = gridDim.x * blockDim.x;
    const int tid = blockIdx.x * blockDim.x + threadIdx.x;
    for (int i = tid; i < 48 * 448; i += stride) {
        int co = i / 448, k = i - co * 448;
        float va = 0.0f, vb = 0.0f;
        if (k < 432) {
            int tap = k / 48, ci = k - tap * 48;
            int dy = tap / 3, dx = tap - dy * 3;
            int wi = ((co * 48 + ci) * 3 + dy) * 3 + dx;
            va = w1_off[wi];
            vb = w1_mod[wi];
        }
        bg1a[i] = (__bf16)va;
        bg1b[i] = (__bf16)vb;
    }
    for (int i = tid; i < 16 * 64; i += stride) {
        int n = i / 64, k = i - n * 64;
        bg2a[i] = (__bf16)((n < 4 && k < 48) ? w2_off[n * 48 + k] : 0.0f);
    }
    for (int i = tid; i < 48 * 64; i += stride) {
        int n = i / 64, k = i - n * 64;
        bg2b[i] = (__bf16)((k < 48) ? w2_mod[n * 48 + k] : 0.0f);
    }
}

// ---------------------------------------------------------------------------
// Kernel 1b: x (NCHW fp32) -> zero-padded NHWC bf16 xb[4][258][258][48].
// Border ring (pad=1) is written as zeros so conv tiles never go OOB and the
// TDM descriptor needs no boundary handling.
// ---------------------------------------------------------------------------
__global__ __launch_bounds__(256) void xcast_kernel(
    const float* __restrict__ x, __bf16* __restrict__ xb) {
    const int total = 4 * 258 * 258;
    int i = blockIdx.x * blockDim.x + threadIdx.x;
    if (i >= total) return;
    int xx = i % 258;
    int t = i / 258;
    int y = t % 258;
    int b = t / 258;
    int gx = xx - 1, gy = y - 1;
    __bf16* o = xb + (long)i * 48;
    if ((unsigned)gx < 256u && (unsigned)gy < 256u) {
        const float* ip = x + ((long)(b * 48) * 256 + gy) * 256 + gx;
        for (int c = 0; c < 48; ++c) o[c] = (__bf16)ip[(long)c * 65536];
    } else {
        for (int c = 0; c < 48; ++c) o[c] = (__bf16)0.0f;
    }
}

// ---------------------------------------------------------------------------
// Kernel 2: integral image.  One block per (b,c) plane of 256x256.
// ---------------------------------------------------------------------------
__global__ __launch_bounds__(256) void integral_kernel(
    const float* __restrict__ x, float* __restrict__ integ) {
    const long plane = (long)blockIdx.x << 16;
    const int t = threadIdx.x;
    {
        const float* xr = x + plane + (long)t * 256;
        float* ir = integ + plane + (long)t * 256;
        float s = 0.0f;
        for (int c = 0; c < 256; ++c) { s += xr[c]; ir[c] = s; }
    }
    __syncthreads();
    {
        float* ic = integ + plane + t;
        float s = 0.0f;
        for (int r = 0; r < 256; ++r) { s += ic[(long)r * 256]; ic[(long)r * 256] = s; }
    }
}

// ---------------------------------------------------------------------------
// Kernel 3: 3x3 conv 48->48 + bias + exact GELU via bf16 WMMA implicit GEMM.
// Block = 256 threads (8 waves), tile = 128 pixels (row strip) x 48 channels.
// gridDim.y selects branch (off / mod).
// x tile (+halo) arrives in LDS as [dy(3)][col(130)][ci(48)] bf16 via the
// Tensor Data Mover (wave 0 issues one 3-D descriptor; TENSORcnt + barrier
// publish it), so A fragments are contiguous 16B ds_load_b128.
// K = tap*48+ci, 14 K-steps of 32.  Output bf16 NHWC.
// ---------------------------------------------------------------------------
__global__ __launch_bounds__(256) void conv3x3_gelu_wmma(
    const __bf16* __restrict__ xb,
    const __bf16* __restrict__ bgA, const __bf16* __restrict__ bgB,
    const float* __restrict__ biasA, const float* __restrict__ biasB,
    __bf16* __restrict__ outA, __bf16* __restrict__ outB) {
    __shared__ __bf16 xs[3 * 130 * 48];   // 37,440 bytes

    const int branch = blockIdx.y;
    const __bf16* __restrict__ bg = branch ? bgB : bgA;
    const float* __restrict__ bias = branch ? biasB : biasA;
    __bf16* __restrict__ outp = branch ? outB : outA;

    const int bid = blockIdx.x;
    const int wt = bid & 1;
    const int h = (bid >> 1) & 255;
    const int b = bid >> 9;
    const int w0 = wt * 128;

    // weights are hot (L2-resident after first touch)
    __builtin_prefetch(bg + (threadIdx.x & 255) * 64, 0, 1);

    // tile start in padded NHWC tensor: rows h..h+2, cols w0..w0+129, ci 0..47
    const long tileBase = ((long)(b * 258 + h) * 258 + w0) * 48;

#if USE_TDM
    if (threadIdx.x < 32) {
        unsigned long long gaddr = (unsigned long long)(uintptr_t)(xb + tileBase);
        unsigned int laddr = (unsigned int)(uintptr_t)(void*)&xs[0];
        u32x4 g0; i32x8 g1; i32x4 g2; i32x4 g3;
        // D# group0: count=1 | lds_addr | global_addr[56:0] | type=2
        g0[0] = 1u;
        g0[1] = laddr;
        g0[2] = (unsigned int)(gaddr & 0xFFFFFFFFull);
        g0[3] = (unsigned int)((gaddr >> 32) & 0x01FFFFFFull) | 0x80000000u;
        // D# group1: data_size=2B; tensor_dim0=48, tensor_dim1=258;
        //            tile = 48 x 130 x 3; strides 48, 258*48=12384
        g1[0] = 0x00010000;               // data_size = 1 (2 bytes)
        g1[1] = (int)(48u << 16);         // tensor_dim0[15:0]
        g1[2] = (int)(258u << 16);        // tensor_dim1[15:0]
        g1[3] = (int)(48u << 16);         // tile_dim0
        g1[4] = (int)((3u << 16) | 130u); // tile_dim2 | tile_dim1
        g1[5] = 48;                       // tensor_dim0_stride[31:0]
        g1[6] = (int)(12384u << 16);      // tensor_dim1_stride[15:0]
        g1[7] = 0;                        // tensor_dim1_stride[47:16]
        // D# group2: tensor_dim2 (never OOB thanks to padded xb)
        g2[0] = 4; g2[1] = 0; g2[2] = 0; g2[3] = 0;
        g3[0] = 0; g3[1] = 0; g3[2] = 0; g3[3] = 0;
#if defined(__clang_major__) && (__clang_major__ >= 23)
        i32x8 g4 = {};
        __builtin_amdgcn_tensor_load_to_lds(g0, g1, g2, g3, g4, 0);
#else
        __builtin_amdgcn_tensor_load_to_lds(g0, g1, g2, g3, 0);
#endif
        __builtin_amdgcn_s_wait_tensorcnt(0);
    }
    __syncthreads();
#else
    // fallback: manual copy (xb tile rows are contiguous per row)
    for (int i = threadIdx.x; i < 3 * 130 * 48; i += 256) {
        int row = i / 6240;           // 130*48
        int rem = i - row * 6240;
        xs[i] = xb[tileBase + (long)row * (258 * 48) + rem];
    }
    __syncthreads();
#endif

    const int lane = threadIdx.x & 31;
    const int wv = threadIdx.x >> 5;
    const int m16 = lane & 15;
    const int hi = lane >> 4;
    const int Mbase = wv * 16;
    const int p = Mbase + m16;   // A-matrix row = pixel within 128-strip

    v8f acc0 = {}, acc1 = {}, acc2 = {};

    for (int kt = 0; kt < 14; ++kt) {
        // A fragment: two contiguous 8-element bf16 loads from LDS
        V16 a;
        {
            int kk = kt * 32 + hi * 16;      // K of V0..V3
            v8bf lo = {};
            if (kk < 432) {
                int tap = kk / 48, ci0 = kk - tap * 48;
                int dy = tap / 3, dx = tap - dy * 3;
                lo = *(const v8bf*)&xs[((dy * 130) + p + dx) * 48 + ci0];
            }
            int kk2 = kk + 8;                // K of V4..V7
            v8bf hv = {};
            if (kk2 < 432) {
                int tap = kk2 / 48, ci0 = kk2 - tap * 48;
                int dy = tap / 3, dx = tap - dy * 3;
                hv = *(const v8bf*)&xs[((dy * 130) + p + dx) * 48 + ci0];
            }
            a.h[0] = lo; a.h[1] = hv;
        }
        // B fragments: lane = column N, 16 contiguous K from global (L2)
        const __bf16* bk = bg + kt * 32 + hi * 16;
        v16bf b0 = *(const v16bf*)(bk + (long)(0 * 16 + m16) * 448);
        v16bf b1 = *(const v16bf*)(bk + (long)(1 * 16 + m16) * 448);
        v16bf b2 = *(const v16bf*)(bk + (long)(2 * 16 + m16) * 448);

        acc0 = __builtin_amdgcn_wmma_f32_16x16x32_bf16(false, a.v, false, b0,
                                                       (short)0, acc0, false, false);
        acc1 = __builtin_amdgcn_wmma_f32_16x16x32_bf16(false, a.v, false, b1,
                                                       (short)0, acc1, false, false);
        acc2 = __builtin_amdgcn_wmma_f32_16x16x32_bf16(false, a.v, false, b2,
                                                       (short)0, acc2, false, false);
    }

    // epilogue: bias + exact GELU, store bf16 NHWC
    const float bb0 = bias[0 * 16 + m16];
    const float bb1 = bias[1 * 16 + m16];
    const float bb2 = bias[2 * 16 + m16];
    for (int r = 0; r < 8; ++r) {
        int pp = Mbase + r + 8 * hi;         // C layout: M = r + 8*hi
        long base = (((long)(b * 256 + h)) * 256 + (w0 + pp)) * 48;
        outp[base + 0 * 16 + m16] = (__bf16)gelu_exact(acc0[r] + bb0);
        outp[base + 1 * 16 + m16] = (__bf16)gelu_exact(acc1[r] + bb1);
        outp[base + 2 * 16 + m16] = (__bf16)gelu_exact(acc2[r] + bb2);
    }
}

// ---------------------------------------------------------------------------
// Kernel 4: fused 1x1 convs (WMMA) + offsets/area + 4-corner bilinear
// sampling of the integral image.  Block = 128 threads (4 waves), 64 pixels.
// ---------------------------------------------------------------------------
__global__ __launch_bounds__(128) void finalize_kernel(
    const __bf16* __restrict__ h1, const __bf16* __restrict__ m1,
    const __bf16* __restrict__ bg2a, const __bf16* __restrict__ bg2b,
    const float* __restrict__ b2_off, const float* __restrict__ b2_mod,
    const float* __restrict__ integ, float* __restrict__ out) {
    __shared__ float subs[64][4];
    __shared__ float scl[64][48];

    const int lane = threadIdx.x & 31;
    const int wv = threadIdx.x >> 5;
    const int m16 = lane & 15;
    const int hi = lane >> 4;
    const int Mbase = wv * 16;
    const long pixBase = (long)blockIdx.x * 64;

    // phase A: WMMA GEMMs, M=16 pixels, K=48 (pad 64)
    {
        const __bf16* aptr = h1 + (pixBase + Mbase + m16) * 48;
        const __bf16* mptr = m1 + (pixBase + Mbase + m16) * 48;
        v8f asub = {}, as0 = {}, as1 = {}, as2 = {};
        for (int kt = 0; kt < 2; ++kt) {
            int kk = kt * 32 + hi * 16;
            int kk2 = kk + 8;
            V16 a, am;
            a.h[0] = (kk < 48) ? *(const v8bf*)(aptr + kk) : (v8bf){};
            a.h[1] = (kk2 < 48) ? *(const v8bf*)(aptr + kk2) : (v8bf){};
            am.h[0] = (kk < 48) ? *(const v8bf*)(mptr + kk) : (v8bf){};
            am.h[1] = (kk2 < 48) ? *(const v8bf*)(mptr + kk2) : (v8bf){};

            v16bf bsub = *(const v16bf*)(bg2a + (long)m16 * 64 + kt * 32 + hi * 16);
            asub = __builtin_amdgcn_wmma_f32_16x16x32_bf16(false, a.v, false, bsub,
                                                           (short)0, asub, false, false);
            const __bf16* bk = bg2b + kt * 32 + hi * 16;
            v16bf b0 = *(const v16bf*)(bk + (long)(0 * 16 + m16) * 64);
            v16bf b1 = *(const v16bf*)(bk + (long)(1 * 16 + m16) * 64);
            v16bf b2 = *(const v16bf*)(bk + (long)(2 * 16 + m16) * 64);
            as0 = __builtin_amdgcn_wmma_f32_16x16x32_bf16(false, am.v, false, b0,
                                                          (short)0, as0, false, false);
            as1 = __builtin_amdgcn_wmma_f32_16x16x32_bf16(false, am.v, false, b1,
                                                          (short)0, as1, false, false);
            as2 = __builtin_amdgcn_wmma_f32_16x16x32_bf16(false, am.v, false, b2,
                                                          (short)0, as2, false, false);
        }
        const float bm0 = b2_mod[0 * 16 + m16];
        const float bm1 = b2_mod[1 * 16 + m16];
        const float bm2 = b2_mod[2 * 16 + m16];
        const float bo = (m16 < 4) ? b2_off[m16] : 0.0f;
        for (int r = 0; r < 8; ++r) {
            int pp = Mbase + r + 8 * hi;
            if (m16 < 4) subs[pp][m16] = asub[r] + bo;
            scl[pp][0 * 16 + m16] = as0[r] + bm0;
            scl[pp][1 * 16 + m16] = as1[r] + bm1;
            scl[pp][2 * 16 + m16] = as2[r] + bm2;
        }
    }
    __syncthreads();

    // phase B: per-pixel offsets, area, 16-tap sampling
    const int t = threadIdx.x;
    const int p = t & 63;
    const int halfsel = t >> 6;           // 2 threads split the 48 channels
    const long idx = pixBase + p;
    const int b = (int)(idx >> 16);
    const int rem = (int)(idx & 65535);
    const int h = rem >> 8;
    const int w = rem & 255;

    const float s0 = subs[p][0], s1 = subs[p][1];
    const float s2 = subs[p][2], s3 = subs[p][3];
    const float dw = 0.5f * (s0 - s1);
    const float dh = 0.5f * (s2 - s3);
    const float mw = fmaxf(fabsf(dw), 0.015625f);   // MIN_PATCH / tp_w
    const float mh = fmaxf(fabsf(dh), 0.015625f);
    float area = mh * mw * 16384.0f;                 // (2mh)(2mw)*128*128/4
    area = fminf(fmaxf(area, 1.0f), 65536.0f);

    const float cx = ((float)w + 0.5f) * (2.0f / 256.0f) - 1.0f;
    const float cy = ((float)h + 0.5f) * (2.0f / 256.0f) - 1.0f;
    const float Xm = fminf(fmaxf(cx - mw * 0.5f, -1.0f), 1.0f);
    const float Xp = fminf(fmaxf(cx + mw * 0.5f, -1.0f), 1.0f);
    const float Ym = fminf(fmaxf(cy - mh * 0.5f, -1.0f), 1.0f);
    const float Yp = fminf(fmaxf(cy + mh * 0.5f, -1.0f), 1.0f);

    // corners: tl(+), tr(-), bl(-), br(+)
    const float Xs[4] = {Xm, Xp, Xm, Xp};
    const float Ys[4] = {Ym, Ym, Yp, Yp};
    const float sgn[4] = {1.0f, -1.0f, -1.0f, 1.0f};
    int off[4][4];
    float wgt[4][4];
    for (int k = 0; k < 4; ++k) {
        float ix = (Xs[k] + 1.0f) * 127.5f;
        float iy = (Ys[k] + 1.0f) * 127.5f;
        float fx0 = floorf(ix), fy0 = floorf(iy);
        float fx = ix - fx0, fy = iy - fy0;
        int x0 = iclamp((int)fx0, 0, 255), x1 = iclamp((int)fx0 + 1, 0, 255);
        int y0 = iclamp((int)fy0, 0, 255), y1 = iclamp((int)fy0 + 1, 0, 255);
        off[k][0] = y0 * 256 + x0; off[k][1] = y0 * 256 + x1;
        off[k][2] = y1 * 256 + x0; off[k][3] = y1 * 256 + x1;
        wgt[k][0] = sgn[k] * (1.0f - fx) * (1.0f - fy);
        wgt[k][1] = sgn[k] * fx * (1.0f - fy);
        wgt[k][2] = sgn[k] * (1.0f - fx) * fy;
        wgt[k][3] = sgn[k] * fx * fy;
    }

    const int co0 = halfsel * 24;
    for (int co = co0; co < co0 + 24; ++co) {
        const float* plane = integ + ((long)(b * 48 + co) << 16);
        float s = 0.0f;
        for (int k = 0; k < 4; ++k) {
            s += plane[off[k][0]] * wgt[k][0] + plane[off[k][1]] * wgt[k][1] +
                 plane[off[k][2]] * wgt[k][2] + plane[off[k][3]] * wgt[k][3];
        }
        out[(((long)(b * 48 + co)) << 16) + h * 256 + w] = s / (area * scl[p][co]);
    }
}

// ---------------------------------------------------------------------------
extern "C" void kernel_launch(void* const* d_in, const int* in_sizes, int n_in,
                              void* d_out, int out_size, void* d_ws, size_t ws_size,
                              hipStream_t stream) {
    const float* x      = (const float*)d_in[0];
    const float* w1_off = (const float*)d_in[1];
    const float* b1_off = (const float*)d_in[2];
    const float* w2_off = (const float*)d_in[3];
    const float* b2_off = (const float*)d_in[4];
    const float* w1_mod = (const float*)d_in[5];
    const float* b1_mod = (const float*)d_in[6];
    const float* w2_mod = (const float*)d_in[7];
    const float* b2_mod = (const float*)d_in[8];

    char* ws = (char*)d_ws;
    float*  integ = (float*)(ws);                          // 50,331,648 B
    __bf16* h1    = (__bf16*)(ws + 50331648);              // 25,165,824 B
    __bf16* m1    = (__bf16*)(ws + 75497472);              // 25,165,824 B
    __bf16* bg1a  = (__bf16*)(ws + 100663296);             // 43,008 B
    __bf16* bg1b  = (__bf16*)(ws + 100706304);             // 43,008 B
    __bf16* bg2a  = (__bf16*)(ws + 100749312);             // 2,048 B
    __bf16* bg2b  = (__bf16*)(ws + 100751360);             // 6,144 B
    __bf16* xb    = (__bf16*)(ws + 100757504);             // 25,560,576 B
    float* out = (float*)d_out;

    wconv_kernel<<<96, 256, 0, stream>>>(w1_off, w1_mod, w2_off, w2_mod,
                                         bg1a, bg1b, bg2a, bg2b);
    xcast_kernel<<<1041, 256, 0, stream>>>(x, xb);
    integral_kernel<<<192, 256, 0, stream>>>(x, integ);
    conv3x3_gelu_wmma<<<dim3(2048, 2), 256, 0, stream>>>(xb, bg1a, bg1b,
                                                         b1_off, b1_mod, h1, m1);
    finalize_kernel<<<4096, 128, 0, stream>>>(h1, m1, bg2a, bg2b,
                                              b2_off, b2_mod, integ, out);
}
